// AmidePredictor_86706799772345
// MI455X (gfx1250) — matrix operand order repaired and checked by reading, not compiled, not run on hardware
//
#include <hip/hip_runtime.h>
#include <hip/hip_bf16.h>

// ---------------------------------------------------------------------------
// AmidePredictor: 3x stacked GAT (N=4096, D=128, L=3) + MLP head, on gfx1250.
// Heavy math (x@W and softmax(P)@h) via v_wmma_f32_16x16x32_bf16.
// Attention h-tiles staged to LDS with async global->LDS loads (ASYNCcnt),
// double-buffered; j-range split 4-ways (exact, since row-max is global).
// ---------------------------------------------------------------------------

#define NN 4096
#define DD 128
#define LL 3
#define HH 385         // 3*128 + 1
#define SS 4           // j-segments for attention parallelism
#define TJ 32          // j-tile (WMMA K)
#define TSTRIDE 40     // padded LDS tile row stride in bf16 elems (80B)

typedef __attribute__((ext_vector_type(16))) __bf16 v16bf;
typedef __attribute__((ext_vector_type(8)))  __bf16 v8bf;
typedef __attribute__((ext_vector_type(8)))  float  v8f;

union ABfrag {
    v16bf v;
    v8bf  h[2];
    unsigned short u[16];
};

__device__ __forceinline__ v8f wmma_bf16(v16bf a, v16bf b, v8f c) {
    return __builtin_amdgcn_wmma_f32_16x16x32_bf16(false, a, false, b, (short)0, c, false, false);
}

__device__ __forceinline__ v8f v8f_zero() {
    v8f z;
#pragma unroll
    for (int i = 0; i < 8; ++i) z[i] = 0.0f;
    return z;
}

// CDNA5 async global->LDS copy (16B per lane), tracked by ASYNCcnt.
__device__ __forceinline__ void async_load_b128(unsigned lds_off, const void* gptr) {
    asm volatile("global_load_async_to_lds_b128 %0, %1, off"
                 :: "v"(lds_off), "v"(gptr) : "memory");
}
__device__ __forceinline__ void wait_async0() {
    asm volatile("s_wait_asynccnt 0x0" ::: "memory");
}

struct GraphIn  { const float *a, *aim, *q, *W, *att; };
struct GraphIns { GraphIn g[3]; };

// ---------------------------------------------------------------------------
// 1) x0 = concat([a, aim, q]) in bf16, row-major [N][D] per graph.
// ---------------------------------------------------------------------------
__global__ void k_concat(GraphIns gis, __bf16* xb) {
    const int g = blockIdx.y;
    const int r = blockIdx.x;
    const int t = threadIdx.x;            // 0..127
    const GraphIn gi = gis.g[g];
    float v;
    if (t < 64)        v = gi.a[r * 64 + t];
    else if (t < 127)  v = gi.aim[r * 63 + (t - 64)];
    else               v = gi.q[r];
    xb[(size_t)g * NN * DD + (size_t)r * DD + t] = (__bf16)v;
}

// ---------------------------------------------------------------------------
// 2) Wt[g,l][n][k] = W[l][k][n]  (bf16)
// ---------------------------------------------------------------------------
__global__ void k_wt(GraphIns gis, __bf16* wt) {
    const int gl = blockIdx.y;            // g*3 + l
    const int gg = gl / 3, ll = gl % 3;
    const int idx = blockIdx.x * blockDim.x + threadIdx.x;
    const int n = idx >> 7, k = idx & 127;
    const float* W = gis.g[gg].W + (size_t)ll * DD * DD;
    wt[(size_t)gl * DD * DD + (size_t)n * DD + k] = (__bf16)W[(size_t)k * DD + n];
}

// ---------------------------------------------------------------------------
// 3) h = x @ W[l]  (bf16 WMMA, f32 acc). Writes h fp32 [N][D] and hT bf16 [D][N].
// ---------------------------------------------------------------------------
__global__ void k_gemm(const __bf16* __restrict__ xb, const __bf16* __restrict__ wt,
                       float* __restrict__ hf, __bf16* __restrict__ ht, int l) {
    const int g    = blockIdx.y;
    const int wv   = threadIdx.x >> 5;
    const int lane = threadIdx.x & 31;
    const int half = lane >> 4;
    const int l16  = lane & 15;
    const int i0   = (blockIdx.x * 8 + wv) * 16;

    const __bf16* X  = xb + (size_t)g * NN * DD;
    const __bf16* Wt = wt + (size_t)(g * 3 + l) * DD * DD;

    v8f acc[8];
#pragma unroll
    for (int nt = 0; nt < 8; ++nt) acc[nt] = v8f_zero();

#pragma unroll
    for (int kt = 0; kt < 4; ++kt) {
        const int k0 = kt * 32;
        ABfrag a;   // A: row i0+l16; elems 0..7 -> K=k0+half*8, 8..15 -> K=k0+16+half*8
        const __bf16* xr = X + (size_t)(i0 + l16) * DD;
        a.h[0] = *(const v8bf*)(xr + k0 + half * 8);
        a.h[1] = *(const v8bf*)(xr + k0 + 16 + half * 8);
#pragma unroll
        for (int nt = 0; nt < 8; ++nt) {
            ABfrag b;   // B from Wt rows: elems 0..7 -> K=k0+half*8, 8..15 -> K=k0+16+half*8
            const __bf16* wr = Wt + (size_t)(nt * 16 + l16) * DD + k0 + half * 8;
            b.h[0] = *(const v8bf*)(wr);
            b.h[1] = *(const v8bf*)(wr + 16);
            acc[nt] = wmma_bf16(a.v, b.v, acc[nt]);
        }
    }

    float*  HF = hf + (size_t)g * NN * DD;
    __bf16* HT = ht + (size_t)g * DD * NN;
#pragma unroll
    for (int nt = 0; nt < 8; ++nt) {
        const int n = nt * 16 + l16;
        v8bf pk;
#pragma unroll
        for (int v = 0; v < 8; ++v) {
            const int row = i0 + half * 8 + v;     // D-frag: VGPR v -> M = v + 8*half
            HF[(size_t)row * DD + n] = acc[nt][v];
            pk[v] = (__bf16)acc[nt][v];
        }
        *(v8bf*)(HT + (size_t)n * NN + i0 + half * 8) = pk;   // transposed 16B store
    }
}

// ---------------------------------------------------------------------------
// 4) s1 = h@a1, s2 = h@a2, s2max = max(s2). One 1024-thread block per graph.
// ---------------------------------------------------------------------------
__global__ void k_svec(GraphIns gis, const float* __restrict__ hf,
                       float* __restrict__ s1, float* __restrict__ s2,
                       float* __restrict__ s2m, int l) {
    const int g = blockIdx.x;
    const int t = threadIdx.x;
    const float* H  = hf + (size_t)g * NN * DD;
    const float* a1 = gis.g[g].att + (size_t)l * 2 * DD;
    const float* a2 = a1 + DD;
    __shared__ float red[1024];

    float mx = -3.0e38f;
    for (int r = t; r < NN; r += 1024) {
        const float* hr = H + (size_t)r * DD;
        float d1 = 0.0f, d2 = 0.0f;
#pragma unroll 4
        for (int k = 0; k < DD; ++k) {
            const float hv = hr[k];
            d1 += hv * a1[k];
            d2 += hv * a2[k];
        }
        s1[g * NN + r] = d1;
        s2[g * NN + r] = d2;
        mx = fmaxf(mx, d2);
    }
    red[t] = mx;
    __syncthreads();
    for (int s = 512; s > 0; s >>= 1) {
        if (t < s) red[t] = fmaxf(red[t], red[t + s]);
        __syncthreads();
    }
    if (t == 0) s2m[g] = red[0];
}

// ---------------------------------------------------------------------------
// 5) Attention partials: for j-segment `seg`, accumulate
//      num[i][d] += sum_j exp(lrelu(s1_i+s2_j)-mrow_i) * h[j][d]   (WMMA)
//      Z[i]      += row sums (WMMA vs all-ones B)
//    mrow_i = lrelu(s1_i + s2max) is GLOBAL -> partials add exactly.
//    h-tiles (32x128 bf16, 8KB) staged to LDS via async loads, double-buffered,
//    shared by all 8 waves of the block.
// ---------------------------------------------------------------------------
__global__ void k_attn(const __bf16* __restrict__ ht,
                       const float* __restrict__ s1, const float* __restrict__ s2,
                       const float* __restrict__ s2m,
                       float* __restrict__ xp, float* __restrict__ zp) {
    __shared__ __align__(16) __bf16 tile[2][DD * TSTRIDE];   // 2 x 10240 B

    const int g    = blockIdx.y;
    const int seg  = blockIdx.z;
    const int tid  = threadIdx.x;
    const int wv   = tid >> 5;
    const int lane = tid & 31;
    const int half = lane >> 4;
    const int l16  = lane & 15;
    const int i0   = (blockIdx.x * 8 + wv) * 16;
    const int jbase = seg * (NN / SS);
    const int NT    = (NN / SS) / TJ;      // 32 tiles per segment

    const __bf16* HT = ht + (size_t)g * DD * NN;
    const float*  S1 = s1 + g * NN;
    const float*  S2 = s2 + g * NN;

    const float s1v  = S1[i0 + l16];
    const float tmx  = s1v + s2m[g];
    const float mrow = (tmx > 0.0f) ? tmx : 0.01f * tmx;

    v8f acc[8];
#pragma unroll
    for (int nt = 0; nt < 8; ++nt) acc[nt] = v8f_zero();
    v8f accz = v8f_zero();

    ABfrag ones;
#pragma unroll
    for (int e = 0; e < 16; ++e) ones.u[e] = 0x3F80;  // bf16 1.0

    // --- stage helper: 512 chunks of 16B, 2 per thread, async to LDS ---
    auto stage = [&](int j0, int buf) {
        for (int c = tid; c < 512; c += 256) {
            const int n = c >> 2, piece = c & 3;
            const __bf16* gsrc = HT + (size_t)n * NN + j0 + piece * 8;
            unsigned lds = (unsigned)(uintptr_t)(&tile[buf][n * TSTRIDE + piece * 8]);
            async_load_b128(lds, (const void*)gsrc);
        }
    };

    stage(jbase, 0);                                   // prologue prefetch

    for (int t = 0; t < NT; ++t) {
        const int cur = t & 1;
        const int j0  = jbase + t * TJ;

        wait_async0();                                 // our slice of tile[cur] done
        __syncthreads();                               // all slices done; prev readers done
        if (t + 1 < NT) stage(jbase + (t + 1) * TJ, cur ^ 1);

        // P tile (16x32 bf16) in registers
        const float4 sa = *(const float4*)(S2 + j0 + half * 8);
        const float4 sb = *(const float4*)(S2 + j0 + half * 8 + 4);
        const float4 sc = *(const float4*)(S2 + j0 + 16 + half * 8);
        const float4 sd = *(const float4*)(S2 + j0 + 16 + half * 8 + 4);
        float sv[16] = { sa.x, sa.y, sa.z, sa.w, sb.x, sb.y, sb.z, sb.w,
                         sc.x, sc.y, sc.z, sc.w, sd.x, sd.y, sd.z, sd.w };
        ABfrag p;
#pragma unroll
        for (int e = 0; e < 16; ++e) {
            float tt = s1v + sv[e];
            tt = (tt > 0.0f) ? tt : 0.01f * tt;        // leaky_relu(0.01)
            p.v[e] = (__bf16)__expf(tt - mrow);        // <= 1
        }

        const __bf16* tb = &tile[cur][0];
#pragma unroll
        for (int nt = 0; nt < 8; ++nt) {
            ABfrag b;   // B[k][n]=h[j0+k][n]: LDS row n, contig k; ds_load_b128 x2
            const __bf16* hr = tb + (nt * 16 + l16) * TSTRIDE + half * 8;
            b.h[0] = *(const v8bf*)(hr);
            b.h[1] = *(const v8bf*)(hr + 16);
            acc[nt] = wmma_bf16(p.v, b.v, acc[nt]);
        }
        accz = wmma_bf16(p.v, ones.v, accz);
    }

    // write partials: xp[(g*SS+seg)][row][d], zp[(g*SS+seg)][row]
    float* XP = xp + (size_t)(g * SS + seg) * NN * DD;
    float* ZP = zp + (size_t)(g * SS + seg) * NN;
#pragma unroll
    for (int v = 0; v < 8; ++v) {
        const int row = i0 + half * 8 + v;
#pragma unroll
        for (int nt = 0; nt < 8; ++nt)
            XP[(size_t)row * DD + nt * 16 + l16] = acc[nt][v];
        if (l16 == 0) ZP[row] = accz[v];
    }
}

// ---------------------------------------------------------------------------
// 5b) Combine segment partials, normalize, write x fp32 + bf16.
// ---------------------------------------------------------------------------
__global__ void k_attn_reduce(const float* __restrict__ xp, const float* __restrict__ zp,
                              float* __restrict__ xf, __bf16* __restrict__ xb) {
    const int g = blockIdx.y;
    const int r = blockIdx.x;
    const int d = threadIdx.x;            // 0..127
    float num = 0.0f, z = 0.0f;
#pragma unroll
    for (int s = 0; s < SS; ++s) {
        num += xp[((size_t)(g * SS + s) * NN + r) * DD + d];
        z   += zp[(size_t)(g * SS + s) * NN + r];
    }
    const float val = num / z;            // total Z >= 1 (max term contributes 1)
    xf[(size_t)g * NN * DD + (size_t)r * DD + d] = val;
    xb[(size_t)g * NN * DD + (size_t)r * DD + d] = (__bf16)val;
}

// ---------------------------------------------------------------------------
// 6) pooled[g][d] = sum_i x_final[g][i][d]
// ---------------------------------------------------------------------------
__global__ void k_pool(const float* __restrict__ xf, float* __restrict__ pooled) {
    const int g = blockIdx.x;
    const int d = threadIdx.x;
    const float* X = xf + (size_t)g * NN * DD;
    float s = 0.0f;
    for (int r = 0; r < NN; ++r) s += X[(size_t)r * DD + d];
    pooled[g * DD + d] = s;
}

// ---------------------------------------------------------------------------
// 7) MLP head.
// ---------------------------------------------------------------------------
__device__ __forceinline__ float gelu_exact(float x) {
    return 0.5f * x * (1.0f + erff(x * 0.7071067811865476f));
}

__global__ void k_head(const float* __restrict__ pooled, const float* __restrict__ control,
                       const float* __restrict__ Wh0, const float* __restrict__ bh0,
                       const float* __restrict__ Wh1, const float* __restrict__ bh1,
                       const float* __restrict__ Wp,  const float* __restrict__ bp,
                       const float* __restrict__ Wo,  const float* __restrict__ bo,
                       float* __restrict__ out) {
    __shared__ float v[HH];
    __shared__ float y[HH];
    const int t = threadIdx.x;            // 0..511
    for (int i = t; i < HH; i += 512) v[i] = (i < 3 * DD) ? pooled[i] : control[0];
    __syncthreads();

    if (t < HH) {
        float s = bh0[t];
        for (int k = 0; k < HH; ++k) s += v[k] * Wh0[k * HH + t];
        y[t] = gelu_exact(s);
    }
    __syncthreads();
    if (t < HH) {
        float s = bh1[t];
        for (int k = 0; k < HH; ++k) s += y[k] * Wh1[k * HH + t];
        v[t] = gelu_exact(s);
    }
    __syncthreads();
    if (t < DD) {
        float s = bp[t];
        for (int k = 0; k < HH; ++k) s += v[k] * Wp[k * DD + t];
        y[t] = gelu_exact(s);
    }
    __syncthreads();
    if (t == 0) {
        float s = bo[0];
        for (int k = 0; k < DD; ++k) s += y[k] * Wo[k];
        out[0] = s;
    }
}

// ---------------------------------------------------------------------------
// Launch
// ---------------------------------------------------------------------------
extern "C" void kernel_launch(void* const* d_in, const int* in_sizes, int n_in,
                              void* d_out, int out_size, void* d_ws, size_t ws_size,
                              hipStream_t stream) {
    (void)in_sizes; (void)n_in; (void)out_size; (void)ws_size;

    GraphIns gis;
    for (int g = 0; g < 3; ++g) {
        const int base = g * 5;
        gis.g[g].a   = (const float*)d_in[base + 0];
        gis.g[g].aim = (const float*)d_in[base + 1];
        gis.g[g].q   = (const float*)d_in[base + 2];
        gis.g[g].W   = (const float*)d_in[base + 3];
        gis.g[g].att = (const float*)d_in[base + 4];
    }
    const float* control = (const float*)d_in[15];
    const float* Wh0 = (const float*)d_in[16];
    const float* bh0 = (const float*)d_in[17];
    const float* Wh1 = (const float*)d_in[18];
    const float* bh1 = (const float*)d_in[19];
    const float* Wp  = (const float*)d_in[20];
    const float* bp  = (const float*)d_in[21];
    const float* Wo  = (const float*)d_in[22];
    const float* bo  = (const float*)d_in[23];

    char* ws = (char*)d_ws;
    auto al = [](size_t x) { return (x + 255) & ~(size_t)255; };
    size_t off = 0;
    __bf16* xb = (__bf16*)(ws + off); off = al(off + (size_t)3 * NN * DD * 2);
    float*  xf = (float*) (ws + off); off = al(off + (size_t)3 * NN * DD * 4);
    __bf16* ht = (__bf16*)(ws + off); off = al(off + (size_t)3 * DD * NN * 2);
    float*  hf = (float*) (ws + off); off = al(off + (size_t)3 * NN * DD * 4);
    __bf16* wt = (__bf16*)(ws + off); off = al(off + (size_t)9 * DD * DD * 2);
    float*  s1 = (float*) (ws + off); off = al(off + (size_t)3 * NN * 4);
    float*  s2 = (float*) (ws + off); off = al(off + (size_t)3 * NN * 4);
    float* s2m = (float*) (ws + off); off = al(off + (size_t)3 * 4);
    float* pooled = (float*)(ws + off); off = al(off + (size_t)3 * DD * 4);
    float*  xp = (float*) (ws + off); off = al(off + (size_t)3 * SS * NN * DD * 4);
    float*  zp = (float*) (ws + off); off = al(off + (size_t)3 * SS * NN * 4);

    k_concat<<<dim3(NN, 3), 128, 0, stream>>>(gis, xb);
    k_wt<<<dim3((DD * DD) / 256, 9), 256, 0, stream>>>(gis, wt);

    for (int l = 0; l < LL; ++l) {
        k_gemm<<<dim3(NN / 16 / 8, 3), 256, 0, stream>>>(xb, wt, hf, ht, l);
        k_svec<<<dim3(3), 1024, 0, stream>>>(gis, hf, s1, s2, s2m, l);
        k_attn<<<dim3(NN / 16 / 8, 3, SS), 256, 0, stream>>>(ht, s1, s2, s2m, xp, zp);
        k_attn_reduce<<<dim3(NN, 3), 128, 0, stream>>>(xp, zp, xf, xb);
    }

    k_pool<<<dim3(3), DD, 0, stream>>>(xf, pooled);
    k_head<<<dim3(1), 512, 0, stream>>>(pooled, control, Wh0, bh0, Wh1, bh1,
                                        Wp, bp, Wo, bo, (float*)d_out);
}
